// GNN2_18940805775493
// MI455X (gfx1250) — matrix operand order repaired and checked by reading, not compiled, not run on hardware
//
#include <hip/hip_runtime.h>

typedef __attribute__((ext_vector_type(16))) _Float16 v16h;
typedef __attribute__((ext_vector_type(8)))  _Float16 v8h;
typedef __attribute__((ext_vector_type(8)))  float    v8f;

#define BLOCK 128
#define WPB   4      // waves (graphs) per block, wave32

__device__ __forceinline__ float lrelu(float x) { return x > 0.f ? x : 0.2f * x; }

__device__ __forceinline__ v8f wmma16(v16h a, v16h b, v8f c) {
  return __builtin_amdgcn_wmma_f32_16x16x32_f16(false, a, false, b, (short)0, c, false, false);
}

// A fragment: rows m0..m0+15, cols k0..k0+31 of row-major f16 matrix, stride ld (halves).
// lane m = l&15; lanes 0-15 hold K {0..7,16..23}, lanes 16-31 hold K {8..15,24..31}.
__device__ __forceinline__ v16h load_frag_a(const _Float16* M, int m0, int k0, int ld, int lane) {
  const _Float16* row = M + (m0 + (lane & 15)) * ld + k0 + ((lane >> 4) << 3);
  v8h lo = *reinterpret_cast<const v8h*>(row);        // K +0..7
  v8h hi = *reinterpret_cast<const v8h*>(row + 16);   // K +16..23
  return __builtin_shufflevector(lo, hi, 0, 1, 2, 3, 4, 5, 6, 7,
                                 8, 9, 10, 11, 12, 13, 14, 15);
}

// B fragment from K-innermost ("B^T") storage: BT[n][k], stride ldk (halves).
__device__ __forceinline__ v16h load_frag_bt(const _Float16* BT, int k0, int n0, int ldk, int lane) {
  const _Float16* p = BT + (n0 + (lane & 15)) * ldk + k0 + ((lane >> 4) << 3);
  v8h lo = *reinterpret_cast<const v8h*>(p);
  v8h hi = *reinterpret_cast<const v8h*>(p + 16);
  return __builtin_shufflevector(lo, hi, 0, 1, 2, 3, 4, 5, 6, 7, 8, 9, 10, 11, 12, 13, 14, 15);
}

__device__ __forceinline__ v16h load_frag_bt_masked(const _Float16* BT, int k0, int n0, int ldk,
                                                    int lane, int nvalid) {
  int n = lane & 15;
  if (n < nvalid)
    return load_frag_bt(BT, k0, n0, ldk, lane);
  v16h z = {};
  return z;
}

// HT[n][m] = (X @ W)[m][n] + bias[n]  (transposed store, node-innermost),
// plus fused attention-projection tile: SB[n][m] = (X @ [W*a_src | W*a_dst])[m][n], n<8.
// W and the projection tile given K-innermost (stride LDK).
template <int NT, int KT, int LDK>
__device__ __forceinline__ void gemm_layer(const _Float16* X, const _Float16* WT,
                                           const _Float16* ST, const float* bias,
                                           _Float16* HT, float* SB, int lane) {
  int col = lane & 15, rh = lane >> 4;
#pragma unroll
  for (int mt = 0; mt < 2; ++mt) {
    v16h a[KT];
#pragma unroll
    for (int kt = 0; kt < KT; ++kt) a[kt] = load_frag_a(X, mt * 16, kt * 32, 128, lane);
#pragma unroll
    for (int nt = 0; nt < NT; ++nt) {
      v8f acc = {};
#pragma unroll
      for (int kt = 0; kt < KT; ++kt) {
        v16h b = load_frag_bt(WT, kt * 32, nt * 16, LDK, lane);
        acc = wmma16(a[kt], b, acc);
      }
      float bv = bias[nt * 16 + col];
      v8h h;
#pragma unroll
      for (int v = 0; v < 8; ++v) h[v] = (_Float16)(acc[v] + bv);
      *reinterpret_cast<v8h*>(HT + (nt * 16 + col) * 32 + mt * 16 + (rh << 3)) = h;
    }
    // fused s-tile: cols 0..3 = s_src per head, 4..7 = s_dst per head (rows 8..15 of ST zero)
    v8f accS = {};
#pragma unroll
    for (int kt = 0; kt < KT; ++kt) {
      v16h b = load_frag_bt(ST, kt * 32, 0, LDK, lane);
      accS = wmma16(a[kt], b, accS);
    }
    if (col < 8)
      *reinterpret_cast<v8f*>(SB + col * 32 + mt * 16 + (rh << 3)) = accS;
  }
}

// GAT attention: X[i, head*D+d] = (1/sum_i) * sum_j exp(lrelu(sd_i+ss_j)-rowmax) * H[j, head*D+d]
// HT feature-major [128][32]; SB[8][32] = {ss[4],sd[4]} x node; X output row-major (next A).
template <int D>
__device__ __forceinline__ void attention(const _Float16* HT, const float* SB, _Float16* X,
                                          _Float16* Al, float* RS, int lane) {
  int col = lane & 15, rh = lane >> 4;
#pragma unroll 1
  for (int head = 0; head < 4; ++head) {
    float sd = SB[(4 + head) * 32 + lane];
    float ss = SB[head * 32 + lane];
    v8f ssv[4];
#pragma unroll
    for (int jo = 0; jo < 4; ++jo)          // uniform-address broadcast loads of all ss_j
      ssv[jo] = *reinterpret_cast<const v8f*>(SB + head * 32 + jo * 8);
    // lrelu monotonic => row max = lrelu(sd_i + max_j ss_j)
    float ssmax = ss;
#pragma unroll
    for (int m = 16; m >= 1; m >>= 1) ssmax = fmaxf(ssmax, __shfl_xor(ssmax, m, 32));
    float rowmax = lrelu(sd + ssmax);
    float sum = 0.f;
#pragma unroll
    for (int jo = 0; jo < 4; ++jo) {        // packed alpha stores: 4 x v8h per head
      v8h pk;
#pragma unroll
      for (int ji = 0; ji < 8; ++ji) {
        float e  = lrelu(sd + ssv[jo][ji]);
        float pv = __expf(e - rowmax);
        sum += pv;
        pk[ji] = (_Float16)pv;
      }
      *reinterpret_cast<v8h*>(Al + lane * 32 + jo * 8) = pk;
    }
    RS[lane] = 1.f / sum;                   // softmax denom folded into D scaling
    __syncthreads();

    constexpr int NTILE = (D == 32) ? 2 : 1;
    constexpr int NV    = (D == 32) ? 16 : 6;
    v16h a0 = load_frag_a(Al, 0, 0, 32, lane);
    v16h a1 = load_frag_a(Al, 16, 0, 32, lane);
#pragma unroll
    for (int nt = 0; nt < NTILE; ++nt) {
      v16h b = (D == 32) ? load_frag_bt(HT, 0, head * D + nt * 16, 32, lane)
                         : load_frag_bt_masked(HT, 0, head * D, 32, lane, NV);
      v8f acc0 = {}, acc1 = {};
      acc0 = wmma16(a0, b, acc0);
      acc1 = wmma16(a1, b, acc1);
#pragma unroll
      for (int v = 0; v < 8; ++v) {
        int r0 = (rh << 3) + v, r1 = 16 + r0;
        if (col < NV) {
          X[r0 * 128 + head * D + nt * 16 + col] = (_Float16)(acc0[v] * RS[r0]);
          X[r1 * 128 + head * D + nt * 16 + col] = (_Float16)(acc1[v] * RS[r1]);
        }
      }
    }
    __syncthreads();
  }
}

__global__ __launch_bounds__(BLOCK) void gat3_wmma_kernel(
    const float* __restrict__ xs, const float* __restrict__ pe,
    const float* __restrict__ W1, const float* __restrict__ as1, const float* __restrict__ ad1, const float* __restrict__ b1,
    const float* __restrict__ W2, const float* __restrict__ as2, const float* __restrict__ ad2, const float* __restrict__ b2,
    const float* __restrict__ W3, const float* __restrict__ as3, const float* __restrict__ ad3, const float* __restrict__ b3,
    const float* __restrict__ lw, const float* __restrict__ lb,
    float* __restrict__ out, int R) {
  // ---- shared weights, all K-innermost (transposed), f16 ----
  __shared__ __align__(32) _Float16 sW1T[128 * 32];   // [out][in p32], k 16..31 zero
  __shared__ __align__(32) _Float16 sW2T[128 * 128];  // [out][in]
  __shared__ __align__(32) _Float16 sW3T[32 * 128];   // [out p32][in], rows 24..31 zero
  __shared__ __align__(32) _Float16 sLinT[64 * 32];   // [out][in p32], k 24..31 zero
  __shared__ __align__(32) _Float16 sS1T[16 * 32];    // [8 proj + 8 zero][in p32]
  __shared__ __align__(32) _Float16 sS2T[16 * 128];
  __shared__ __align__(32) _Float16 sS3T[16 * 128];
  __shared__ float sB1[128], sB2[128], sB3[32], sLinB[64];
  // ---- per-wave graph buffers ----
  __shared__ __align__(32) _Float16 sX[WPB][32 * 128];   // node-major (A operand)
  __shared__ __align__(32) _Float16 sHT[WPB][128 * 32];  // feature-major (B operand)
  __shared__ __align__(32) _Float16 sAl[WPB][32 * 32];   // alpha, target-row-major
  __shared__ __align__(32) float sSB[WPB][8 * 32];       // [ss(4)|sd(4)] x node, f32
  __shared__ float sRS[WPB][32];

  int tid = threadIdx.x, lane = tid & 31, wv = tid >> 5;

  // ---- stage weights f32 -> f16 LDS, transposed (one-time, block cooperative) ----
  for (int i = tid; i < 128 * 32; i += BLOCK) {
    int o = i >> 5, k = i & 31;
    sW1T[i] = (k < 16) ? (_Float16)W1[k * 128 + o] : (_Float16)0.f;
  }
  for (int i = tid; i < 128 * 128; i += BLOCK) {
    int o = i >> 7, k = i & 127;
    sW2T[i] = (_Float16)W2[k * 128 + o];
  }
  for (int i = tid; i < 32 * 128; i += BLOCK) {
    int o = i >> 7, k = i & 127;
    sW3T[i] = (o < 24) ? (_Float16)W3[k * 24 + o] : (_Float16)0.f;
  }
  for (int i = tid; i < 64 * 32; i += BLOCK) {
    int o = i >> 5, k = i & 31;
    sLinT[i] = (k < 24) ? (_Float16)lw[k * 64 + o] : (_Float16)0.f;
  }
  // attention projection tiles: row n<4 -> W*a_src[head n]; n in 4..7 -> W*a_dst[head n-4]
  for (int i = tid; i < 16 * 32; i += BLOCK) {
    int n = i >> 5, k = i & 31;
    float s = 0.f;
    if (n < 8 && k < 16) {
      const float* av = (n < 4) ? as1 : ad1;
      int h = n & 3;
      for (int d = 0; d < 32; ++d) s += W1[k * 128 + h * 32 + d] * av[h * 32 + d];
    }
    sS1T[i] = (_Float16)s;
  }
  for (int i = tid; i < 16 * 128; i += BLOCK) {
    int n = i >> 7, k = i & 127;
    float s = 0.f;
    if (n < 8) {
      const float* av = (n < 4) ? as2 : ad2;
      int h = n & 3;
      for (int d = 0; d < 32; ++d) s += W2[k * 128 + h * 32 + d] * av[h * 32 + d];
    }
    sS2T[i] = (_Float16)s;
  }
  for (int i = tid; i < 16 * 128; i += BLOCK) {
    int n = i >> 7, k = i & 127;
    float s = 0.f;
    if (n < 8) {
      const float* av = (n < 4) ? as3 : ad3;
      int h = n & 3;
      for (int d = 0; d < 6; ++d) s += W3[k * 24 + h * 6 + d] * av[h * 6 + d];
    }
    sS3T[i] = (_Float16)s;
  }
  if (tid < 128) { sB1[tid] = b1[tid]; sB2[tid] = b2[tid]; }
  if (tid < 32)  sB3[tid] = (tid < 24) ? b3[tid] : 0.f;
  if (tid < 64)  sLinB[tid] = lb[tid];

  // ---- stage this wave's graph: node feature = [cell, 15 pos-enc], K-padded to 32 ----
  int g = blockIdx.x * WPB + wv;
  int bIdx = g / R, rIdx = g - bIdx * R;
  {
    _Float16* X = sX[wv];
    float xv = xs[((size_t)bIdx * R + rIdx) * 32 + lane];
    const float* p = pe + ((size_t)bIdx * 32 + lane) * 15;
    v8h q0, q1, z = {};
    q0[0] = (_Float16)xv;
#pragma unroll
    for (int e = 0; e < 7; ++e) q0[1 + e] = (_Float16)p[e];
#pragma unroll
    for (int e = 7; e < 15; ++e) q1[e - 7] = (_Float16)p[e];
    *reinterpret_cast<v8h*>(X + lane * 128 + 0)  = q0;
    *reinterpret_cast<v8h*>(X + lane * 128 + 8)  = q1;
    *reinterpret_cast<v8h*>(X + lane * 128 + 16) = z;
    *reinterpret_cast<v8h*>(X + lane * 128 + 24) = z;
  }
  __syncthreads();

  _Float16* X = sX[wv]; _Float16* HT = sHT[wv]; _Float16* Al = sAl[wv];
  float* SB = sSB[wv];  float* RS = sRS[wv];

  // Layer 1
  gemm_layer<8, 1, 32>(X, sW1T, sS1T, sB1, HT, SB, lane);
  __syncthreads();
  attention<32>(HT, SB, X, Al, RS, lane);
  // Layer 2
  gemm_layer<8, 4, 128>(X, sW2T, sS2T, sB2, HT, SB, lane);
  __syncthreads();
  attention<32>(HT, SB, X, Al, RS, lane);
  // Layer 3
  gemm_layer<2, 4, 128>(X, sW3T, sS3T, sB3, HT, SB, lane);
  __syncthreads();
  attention<6>(HT, SB, X, Al, RS, lane);

  // zero K-pad cols 24..31 of X before final linear (stale layer-2 data there)
  {
    v8h z = {};
    *reinterpret_cast<v8h*>(X + lane * 128 + 24) = z;
  }
  __syncthreads();

  // Final: (X[32x24p32] @ lin_w).mean(rows) + lin_b ; both M-tiles summed into one acc
  {
    v16h a0 = load_frag_a(X, 0, 0, 128, lane);
    v16h a1 = load_frag_a(X, 16, 0, 128, lane);
#pragma unroll
    for (int nt = 0; nt < 4; ++nt) {
      v16h b = load_frag_bt(sLinT, 0, nt * 16, 32, lane);
      v8f acc = {};
      acc = wmma16(a0, b, acc);
      acc = wmma16(a1, b, acc);
      float cs = 0.f;
#pragma unroll
      for (int v = 0; v < 8; ++v) cs += acc[v];   // rows held by this lane
      cs += __shfl_xor(cs, 16, 32);               // combine row halves (same column)
      if (lane < 16)
        out[(size_t)g * 64 + nt * 16 + lane] = cs * (1.f / 32.f) + sLinB[nt * 16 + lane];
    }
  }
}

extern "C" void kernel_launch(void* const* d_in, const int* in_sizes, int n_in,
                              void* d_out, int out_size, void* d_ws, size_t ws_size,
                              hipStream_t stream) {
  const float* xs  = (const float*)d_in[0];
  const float* pe  = (const float*)d_in[1];
  const float* W1  = (const float*)d_in[2];
  const float* as1 = (const float*)d_in[3];
  const float* ad1 = (const float*)d_in[4];
  const float* b1  = (const float*)d_in[5];
  const float* W2  = (const float*)d_in[6];
  const float* as2 = (const float*)d_in[7];
  const float* ad2 = (const float*)d_in[8];
  const float* b2  = (const float*)d_in[9];
  const float* W3  = (const float*)d_in[10];
  const float* as3 = (const float*)d_in[11];
  const float* ad3 = (const float*)d_in[12];
  const float* b3  = (const float*)d_in[13];
  const float* lw  = (const float*)d_in[14];
  const float* lb  = (const float*)d_in[15];
  float* out = (float*)d_out;

  int G = in_sizes[0] / 32;            // B*R graphs (16384)
  int B = in_sizes[1] / (32 * 15);     // 4
  int R = G / B;                       // 4096

  dim3 grid(G / WPB), block(BLOCK);
  gat3_wmma_kernel<<<grid, block, 0, stream>>>(xs, pe, W1, as1, ad1, b1,
                                               W2, as2, ad2, b2,
                                               W3, as3, ad3, b3,
                                               lw, lb, out, R);
}
